// DeepEMDSimple_10668698763880
// MI455X (gfx1250) — matrix operand order repaired and checked by reading, not compiled, not run on hardware
//
#include <hip/hip_runtime.h>
#include <stdint.h>
#include <stddef.h>

// ---------------------------------------------------------------------------
// CDNA5 (gfx1250) wave32 WMMA bf16 implementation of DeepEMD-simple forward.
// All heavy contractions go through v_wmma_f32_16x16x32_bf16.
// ---------------------------------------------------------------------------

typedef __attribute__((ext_vector_type(16))) __bf16         v16bf;
typedef __attribute__((ext_vector_type(8)))  float          v8f;
typedef __attribute__((ext_vector_type(8)))  unsigned short v8us;
typedef __attribute__((ext_vector_type(16))) unsigned short v16us;

#define DEV __device__ __forceinline__

DEV unsigned short f2bf(float f) {
  union { float f; unsigned int u; } c; c.f = f;
  unsigned int u = c.u;
  unsigned int r = u + 0x7FFFu + ((u >> 16) & 1u);   // round-to-nearest-even
  return (unsigned short)(r >> 16);
}

// A-fragment (16x32 bf16, MxK), row-major source, row = lane&15.
// Lanes 0-15: K {klo=0: 0..15}; lanes 16-31: K {klo=8: 16..31}.
DEV v16bf load_a_frag(const unsigned short* rowPtr, int klo) {
  v8us lo = *(const v8us*)(rowPtr + 2 * klo);
  v8us hi = *(const v8us*)(rowPtr + 2 * klo + 8);
  v16us r;
#pragma unroll
  for (int i = 0; i < 8; ++i) { r[i] = lo[i]; r[i + 8] = hi[i]; }
  return __builtin_bit_cast(v16bf, r);
}

// B-fragment (32x16 bf16, KxN), column n = lane&15 stored K-contiguous;
// lanes 0-15 carry K 0..15, lanes 16-31 carry K 16..31 (caller adds half*16).
DEV v16bf load_b_frag(const unsigned short* p) {
  return __builtin_bit_cast(v16bf, *(const v16us*)p);
}

DEV v8f wmma_bf16(v16bf a, v16bf b, v8f c) {
  return __builtin_amdgcn_wmma_f32_16x16x32_bf16(false, a, false, b,
                                                 (short)0, c, false, false);
}

// ---------------------------------------------------------------------------
// Weight repack: (Cout, Cin, 3, 3) f32 -> (Cout, Kpad) bf16, K = Cin*9.
// ---------------------------------------------------------------------------
__global__ void wconvert_kernel(const float* __restrict__ w,
                                unsigned short* __restrict__ out,
                                int K, int Kpad, int n) {
  int idx = blockIdx.x * blockDim.x + threadIdx.x;
  if (idx >= n) return;
  int k = idx % Kpad;
  int cout = idx / Kpad;
  float v = (k < K) ? w[(size_t)cout * K + k] : 0.f;
  out[idx] = f2bf(v);
}

// ---------------------------------------------------------------------------
// im2col for conv1: reads query/support f32 directly (3ch, 84x84, SAME 3x3),
// emits bf16 column-major-K matrix (Kpad=32, ldcols=7056) per image.
// ---------------------------------------------------------------------------
__global__ void im2col1_kernel(const float* __restrict__ query,
                               const float* __restrict__ support,
                               unsigned short* __restrict__ colbuf,
                               int imgBase, int nImgs) {
  int idx = blockIdx.x * blockDim.x + threadIdx.x;
  int total = nImgs * 7056 * 32;
  if (idx >= total) return;
  int k   = idx & 31;
  int col = (idx >> 5) % 7056;
  int il  = (idx >> 5) / 7056;
  int g = imgBase + il;
  const float* src = (g < 150) ? (query + (size_t)g * 3 * 7056)
                               : (support + (size_t)(g - 150) * 3 * 7056);
  float v = 0.f;
  if (k < 27) {
    int c = k / 9, rem = k % 9;
    int ky = rem / 3, kx = rem % 3;
    int y = col / 84, x = col % 84;
    int iy = y + ky - 1, ix = x + kx - 1;
    if (iy >= 0 && iy < 84 && ix >= 0 && ix < 84)
      v = src[(size_t)c * 7056 + iy * 84 + ix];
  }
  colbuf[((size_t)il * 7056 + col) * 32 + k] = f2bf(v);
}

// ---------------------------------------------------------------------------
// Generic im2col, Cin=64 (Kpad = K = 576). feat layout: (img, 64, H, W) f32.
// Columns >= H*W are zero-padded so WMMA B loads are always defined.
// ---------------------------------------------------------------------------
__global__ void im2col64_kernel(const float* __restrict__ feat,
                                unsigned short* __restrict__ colbuf,
                                int H, int W, int ldcols, int nImgs) {
  int idx = blockIdx.x * blockDim.x + threadIdx.x;
  int total = nImgs * ldcols * 576;
  if (idx >= total) return;
  int k   = idx % 576;
  int col = (idx / 576) % ldcols;
  int il  = idx / (576 * ldcols);
  int HW = H * W;
  float v = 0.f;
  if (col < HW) {
    int c = k / 9, rem = k % 9;
    int ky = rem / 3, kx = rem % 3;
    int y = col / W, x = col % W;
    int iy = y + ky - 1, ix = x + kx - 1;
    if (iy >= 0 && iy < H && ix >= 0 && ix < W)
      v = feat[((size_t)il * 64 + c) * HW + iy * W + ix];
  }
  colbuf[((size_t)il * ldcols + col) * 576 + k] = f2bf(v);
}

// ---------------------------------------------------------------------------
// Conv as GEMM: out[img][cout][col] = relu(scale*acc + bias),
// acc = sum_k Wmat[cout,k] * colbuf[img][col][k].  M=64 -> 4 waves per block.
// ---------------------------------------------------------------------------
__global__ __launch_bounds__(128) void gemm_conv_kernel(
    const unsigned short* __restrict__ Wmat,    // 64 x Kpad, row-major
    const unsigned short* __restrict__ colbuf,  // img x ldcols x Kpad (K contig)
    const float* __restrict__ scale, const float* __restrict__ bias,
    float* __restrict__ out,                    // (img, 64, HW)
    int Kpad, int HW, int ldcols) {
  const int lane = threadIdx.x & 31;
  const int wave = threadIdx.x >> 5;   // cout tile 0..3
  const int ctile = blockIdx.x;        // column tile
  const int img = blockIdx.y;
  const int m = lane & 15;
  const int half = lane >> 4;
  const int klo = half * 8;

  const unsigned short* wrow = Wmat + (size_t)(wave * 16 + m) * Kpad;
  const unsigned short* bcol = colbuf + (size_t)img * ldcols * Kpad
                             + (size_t)(ctile * 16 + m) * Kpad + half * 16;
  v8f acc = {};
  for (int k = 0; k < Kpad; k += 32) {
    v16bf a = load_a_frag(wrow + k, klo);
    v16bf b = load_b_frag(bcol + k);
    acc = wmma_bf16(a, b, acc);
  }
  const int col = ctile * 16 + m;      // D-frag: N = lane&15
  float* obase = out + (size_t)img * 64 * HW;
#pragma unroll
  for (int r = 0; r < 8; ++r) {        // D-frag: M = r + 8*half
    int cout = wave * 16 + r + half * 8;
    if (col < HW) {
      float v = acc[r] * scale[cout] + bias[cout];
      obase[(size_t)cout * HW + col] = v > 0.f ? v : 0.f;
    }
  }
}

// ---------------------------------------------------------------------------
// 2x2 max-pool, stride 2. in: (img,64,2Ho,2Wo), out: (img,64,Ho,Wo).
// ---------------------------------------------------------------------------
__global__ void pool2x2_kernel(const float* __restrict__ in,
                               float* __restrict__ out,
                               int n, int Ho, int Wo) {
  int idx = blockIdx.x * blockDim.x + threadIdx.x;
  if (idx >= n) return;
  int x = idx % Wo;
  int y = (idx / Wo) % Ho;
  int rc = idx / (Wo * Ho);
  int Wi = Wo * 2, Hi = Ho * 2;
  const float* p = in + (size_t)rc * Hi * Wi + (size_t)(2 * y) * Wi + 2 * x;
  out[idx] = fmaxf(fmaxf(p[0], p[1]), fmaxf(p[Wi], p[Wi + 1]));
}

// ---------------------------------------------------------------------------
// Query L2-norm over channels -> bf16 descriptor-major layout [img][col][64],
// columns padded to 448 (zero pad so WMMA A loads are defined).
// ---------------------------------------------------------------------------
__global__ void qnorm_kernel(const float* __restrict__ feat,   // (150,64,441)
                             unsigned short* __restrict__ qn) {
  int idx = blockIdx.x * blockDim.x + threadIdx.x;
  if (idx >= 150 * 448) return;
  int col = idx % 448;
  int img = idx / 448;
  unsigned short* o = qn + ((size_t)img * 448 + col) * 64;
  if (col >= 441) {
#pragma unroll
    for (int d = 0; d < 64; ++d) o[d] = 0;
    return;
  }
  const float* f = feat + (size_t)img * 64 * 441 + col;
  float ss = 0.f;
  for (int d = 0; d < 64; ++d) { float v = f[(size_t)d * 441]; ss += v * v; }
  float inv = 1.f / fmaxf(sqrtf(ss), 1e-12f);
  for (int d = 0; d < 64; ++d) o[d] = f2bf(f[(size_t)d * 441] * inv);
}

// ---------------------------------------------------------------------------
// Support: mean over 5 shots, then L2-norm over channels -> [pair][col][64].
// Support feature images start at global index 150: idx = 150 + b*25 + c*5 + s.
// ---------------------------------------------------------------------------
__global__ void snorm_kernel(const float* __restrict__ feat,
                             unsigned short* __restrict__ sn) {
  int idx = blockIdx.x * blockDim.x + threadIdx.x;
  if (idx >= 10 * 448) return;
  int col = idx % 448;
  int pair = idx / 448;                    // pair = b*5 + way
  int b = pair / 5, c = pair % 5;
  unsigned short* o = sn + ((size_t)pair * 448 + col) * 64;
  if (col >= 441) {
#pragma unroll
    for (int d = 0; d < 64; ++d) o[d] = 0;
    return;
  }
  const float* base = feat + (size_t)(150 + b * 25 + c * 5) * 64 * 441 + col;
  float ss = 0.f;
  for (int d = 0; d < 64; ++d) {
    float m = 0.f;
    for (int s = 0; s < 5; ++s) m += base[((size_t)s * 64 + d) * 441];
    m *= 0.2f;
    ss += m * m;
  }
  float inv = 1.f / fmaxf(sqrtf(ss), 1e-12f);
  for (int d = 0; d < 64; ++d) {
    float m = 0.f;
    for (int s = 0; s < 5; ++s) m += base[((size_t)s * 64 + d) * 441];
    o[d] = f2bf(m * 0.2f * inv);
  }
}

// ---------------------------------------------------------------------------
// Similarity: per (pair p = qimg*5+way, m-tile) wave computes sim = Q^T S
// (K=64, two WMMA steps per 16x16 tile), running max over n with padded
// columns masked to -inf, cross-lane max via shfl_xor, partial row sums out.
// ---------------------------------------------------------------------------
__global__ __launch_bounds__(128) void sim_kernel(
    const unsigned short* __restrict__ qn,   // (150, 448, 64) bf16
    const unsigned short* __restrict__ sn,   // (10, 448, 64) bf16
    float* __restrict__ partials) {          // (750, 28)
  const int lane = threadIdx.x & 31;
  const int wave = threadIdx.x >> 5;
  const int mtile = blockIdx.x * 4 + wave;   // 0..27
  const int p = blockIdx.y;                  // 0..749
  const int qimg = p / 5;
  const int way = p % 5;
  const int b = qimg / 75;

  const unsigned short* Q = qn + (size_t)qimg * 448 * 64;
  const unsigned short* S = sn + (size_t)(b * 5 + way) * 448 * 64;

  const int m = lane & 15;
  const int half = lane >> 4;
  const int klo = half * 8;

  const unsigned short* qrow = Q + (size_t)(mtile * 16 + m) * 64;
  v16bf a0 = load_a_frag(qrow, klo);         // K = 0..31
  v16bf a1 = load_a_frag(qrow + 32, klo);    // K = 32..63

  float runmax[8];
#pragma unroll
  for (int r = 0; r < 8; ++r) runmax[r] = -__builtin_inff();

  const int ncol_off = lane & 15;
  for (int nt = 0; nt < 28; ++nt) {
    const unsigned short* scol = S + (size_t)(nt * 16 + ncol_off) * 64 + half * 16;
    v16bf b0 = load_b_frag(scol);
    v16bf b1 = load_b_frag(scol + 32);
    v8f acc = {};
    acc = wmma_bf16(a0, b0, acc);
    acc = wmma_bf16(a1, b1, acc);
    bool valid = (nt * 16 + ncol_off) < 441;
#pragma unroll
    for (int r = 0; r < 8; ++r) {
      float v = valid ? acc[r] : -__builtin_inff();
      runmax[r] = fmaxf(runmax[r], v);
    }
  }
  // max across the 16 lanes of each half (all n residues covered)
#pragma unroll
  for (int off = 8; off >= 1; off >>= 1) {
#pragma unroll
    for (int r = 0; r < 8; ++r)
      runmax[r] = fmaxf(runmax[r], __shfl_xor(runmax[r], off, 32));
  }
  // sum best over valid query descriptors of this tile
  float s = 0.f;
  int mbase = mtile * 16 + 8 * half;         // D-frag: M = r + 8*half
#pragma unroll
  for (int r = 0; r < 8; ++r)
    if (mbase + r < 441) s += runmax[r];
  s += __shfl_xor(s, 16, 32);                // combine the two halves
  if (lane == 0) partials[(size_t)p * 28 + mtile] = s;
}

__global__ void finalize_kernel(const float* __restrict__ partials,
                                float* __restrict__ out) {
  int p = blockIdx.x * blockDim.x + threadIdx.x;
  if (p >= 750) return;
  float s = 0.f;
  for (int t = 0; t < 28; ++t) s += partials[(size_t)p * 28 + t];
  out[p] = s * (1.f / 441.f);
}

// ---------------------------------------------------------------------------
// Host orchestration.
// ---------------------------------------------------------------------------
extern "C" void kernel_launch(void* const* d_in, const int* in_sizes, int n_in,
                              void* d_out, int out_size, void* d_ws, size_t ws_size,
                              hipStream_t stream) {
  (void)in_sizes; (void)n_in; (void)out_size; (void)ws_size;
  const float* query   = (const float*)d_in[0];
  const float* support = (const float*)d_in[1];
  const float* w1 = (const float*)d_in[2];
  const float* w2 = (const float*)d_in[3];
  const float* w3 = (const float*)d_in[4];
  const float* w4 = (const float*)d_in[5];
  const float* s1 = (const float*)d_in[6];
  const float* b1 = (const float*)d_in[7];
  const float* s2 = (const float*)d_in[8];
  const float* b2 = (const float*)d_in[9];
  const float* s3 = (const float*)d_in[10];
  const float* b3 = (const float*)d_in[11];
  const float* s4 = (const float*)d_in[12];
  const float* b4 = (const float*)d_in[13];

  const int CH = 10;                         // images per chunk (20 chunks)
  char* ws = (char*)d_ws;
  size_t off = 0;
  auto alloc = [&](size_t bytes) -> char* {
    char* pp = ws + off;
    off = (off + bytes + 255) & ~(size_t)255;
    return pp;
  };

  unsigned short* w1m = (unsigned short*)alloc((size_t)64 * 32 * 2);
  unsigned short* w2m = (unsigned short*)alloc((size_t)64 * 576 * 2);
  unsigned short* w3m = (unsigned short*)alloc((size_t)64 * 576 * 2);
  unsigned short* w4m = (unsigned short*)alloc((size_t)64 * 576 * 2);
  float* conv4        = (float*)alloc((size_t)200 * 64 * 441 * 4);
  unsigned short* qn  = (unsigned short*)alloc((size_t)150 * 448 * 64 * 2);
  unsigned short* sn  = (unsigned short*)alloc((size_t)10 * 448 * 64 * 2);
  float* partials     = (float*)alloc((size_t)750 * 28 * 4);
  unsigned short* colbuf = (unsigned short*)alloc((size_t)CH * 1776 * 576 * 2);
  float* F0           = (float*)alloc((size_t)CH * 64 * 7056 * 4);
  float* F1           = (float*)alloc((size_t)CH * 64 * 1764 * 4);

  // weight repack (once per launch — weights are inputs)
  { int n = 64 * 32;  wconvert_kernel<<<(n + 255) / 256, 256, 0, stream>>>(w1, w1m, 27, 32, n); }
  { int n = 64 * 576; wconvert_kernel<<<(n + 255) / 256, 256, 0, stream>>>(w2, w2m, 576, 576, n); }
  { int n = 64 * 576; wconvert_kernel<<<(n + 255) / 256, 256, 0, stream>>>(w3, w3m, 576, 576, n); }
  { int n = 64 * 576; wconvert_kernel<<<(n + 255) / 256, 256, 0, stream>>>(w4, w4m, 576, 576, n); }

  for (int chunk = 0; chunk < 20; ++chunk) {
    int imgBase = chunk * CH;
    // conv1: 84x84, K=27->32
    { int n = CH * 7056 * 32;
      im2col1_kernel<<<(n + 255) / 256, 256, 0, stream>>>(query, support, colbuf, imgBase, CH); }
    gemm_conv_kernel<<<dim3(441, CH), 128, 0, stream>>>(w1m, colbuf, s1, b1, F0, 32, 7056, 7056);
    { int n = CH * 64 * 42 * 42;
      pool2x2_kernel<<<(n + 255) / 256, 256, 0, stream>>>(F0, F1, n, 42, 42); }
    // conv2: 42x42, K=576, cols padded 1764->1776
    { int n = CH * 1776 * 576;
      im2col64_kernel<<<(n + 255) / 256, 256, 0, stream>>>(F1, colbuf, 42, 42, 1776, CH); }
    gemm_conv_kernel<<<dim3(111, CH), 128, 0, stream>>>(w2m, colbuf, s2, b2, F0, 576, 1764, 1776);
    { int n = CH * 64 * 21 * 21;
      pool2x2_kernel<<<(n + 255) / 256, 256, 0, stream>>>(F0, F1, n, 21, 21); }
    // conv3: 21x21, cols padded 441->448
    { int n = CH * 448 * 576;
      im2col64_kernel<<<(n + 255) / 256, 256, 0, stream>>>(F1, colbuf, 21, 21, 448, CH); }
    gemm_conv_kernel<<<dim3(28, CH), 128, 0, stream>>>(w3m, colbuf, s3, b3, F0, 576, 441, 448);
    // conv4 -> persistent store
    { int n = CH * 448 * 576;
      im2col64_kernel<<<(n + 255) / 256, 256, 0, stream>>>(F0, colbuf, 21, 21, 448, CH); }
    gemm_conv_kernel<<<dim3(28, CH), 128, 0, stream>>>(
        w4m, colbuf, s4, b4, conv4 + (size_t)imgBase * 64 * 441, 576, 441, 448);
  }

  { int n = 150 * 448; qnorm_kernel<<<(n + 255) / 256, 256, 0, stream>>>(conv4, qn); }
  { int n = 10 * 448;  snorm_kernel<<<(n + 255) / 256, 256, 0, stream>>>(conv4, sn); }
  sim_kernel<<<dim3(7, 750), 128, 0, stream>>>(qn, sn, partials);
  finalize_kernel<<<(750 + 255) / 256, 256, 0, stream>>>(partials, (float*)d_out);
}
